// HeteGAT_11716670784011
// MI455X (gfx1250) — matrix operand order repaired and checked by reading, not compiled, not run on hardware
//
#include <hip/hip_runtime.h>
#include <hip/hip_bf16.h>

typedef __attribute__((ext_vector_type(16))) _Float16 v16h;
typedef __attribute__((ext_vector_type(2)))  _Float16 half2_t;
typedef __attribute__((ext_vector_type(8)))  float    v8f;
typedef __attribute__((ext_vector_type(4)))  unsigned int v4u;

#define N_H 30000
#define N_O 30000
#define DEG 16
#define HH  4
#define DHD 64
#define DIN 256
#define DEMB 128
#define HID 256
#define NEG_SLOPE 0.01f

#define B_STRIDE 136   // halves per B row in LDS (272B: 16B aligned, bank-spread)
#define A_STRIDE 40    // halves per A row in LDS (80B: 16B aligned, conflict-free)

// ---------------------------------------------------------------------------
// prep: vdst1[h][d] = sum_k W_dst1[h,d,k] * a1[h, 64+k];  zero w-accumulators
// ---------------------------------------------------------------------------
__global__ void prep_kernel(const float* __restrict__ Wdst,
                            const float* __restrict__ a1,
                            float* __restrict__ vdst1,
                            float* __restrict__ wacc) {
  int t = blockIdx.x * blockDim.x + threadIdx.x;
  if (t == 0) { wacc[0] = 0.f; wacc[1] = 0.f; }
  if (t < HH * DIN) {
    int h = t >> 8, d = t & 255;
    float s = 0.f;
    const float* wp = Wdst + (size_t)h * DIN * DHD + (size_t)d * DHD;
    const float* ap = a1 + h * (2 * DHD) + DHD;
#pragma unroll 8
    for (int k = 0; k < DHD; ++k) s += wp[k] * ap[k];
    vdst1[t] = s;
  }
}

// ---------------------------------------------------------------------------
// Shared helpers for the WMMA GEMMs
// ---------------------------------------------------------------------------
// Stage a 64(n) x 256(k) f32 weight stripe, transposed+converted, into LDS f16
// laid out [n][k] with row stride B_STRIDE halves. Bg[(k)*ldb + n] is source.
__device__ __forceinline__ void stage_B(const float* __restrict__ Bg, int ldb,
                                        _Float16* __restrict__ sB) {
  const int t = threadIdx.x;
  const int n = t & 63;
  const int r = t >> 6;           // 0..3
#pragma unroll 4
  for (int i = 0; i < 32; ++i) {
    int dp = i * 4 + r;           // k-pair index 0..127
    float f0 = Bg[(size_t)(2 * dp) * ldb + n];
    float f1 = Bg[(size_t)(2 * dp + 1) * ldb + n];
    half2_t hv; hv.x = (_Float16)f0; hv.y = (_Float16)f1;
    *(half2_t*)&sB[n * B_STRIDE + 2 * dp] = hv;
  }
}

// A fragment (16x32 f16, ISA layout): lane m=lane&15; halves are contiguous:
// e=0..7 -> K = khiA+e ; e=8..15 -> K = 16+khiA+(e-8)  => two b128 LDS loads.
__device__ __forceinline__ v16h load_a_frag(const _Float16* __restrict__ sA,
                                            int lane) {
  const int m = lane & 15;
  const int khiA = (lane >> 4) << 3;   // 0 or 8
  union { v16h h; v4u u[2]; } f;
  f.u[0] = *(const v4u*)(sA + m * A_STRIDE + khiA);
  f.u[1] = *(const v4u*)(sA + m * A_STRIDE + 16 + khiA);
  return f.h;
}

// B fragment (32x16 f16): lane col n, K = kb..kb+7 and kb+16..kb+23 in [n][k] LDS.
__device__ __forceinline__ v16h load_b_frag(const _Float16* __restrict__ sB,
                                            int lane, int ntile, int k0) {
  const int n = ntile * 16 + (lane & 15);
  const int kb = k0 + ((lane >> 4) << 3);  // 0 or 8 within the 32-K step
  union { v16h h; v4u u[2]; } f;
  f.u[0] = *(const v4u*)(sB + n * B_STRIDE + kb);
  f.u[1] = *(const v4u*)(sB + n * B_STRIDE + kb + 16);
  return f.h;
}

// ---------------------------------------------------------------------------
// GAT GEMM:  Z[M,256] = concat(eA,gA)[M,256] @ Wperm,  B[d, h*64+k] = W[h,d,k].
// Block: 8 waves, one 64-col stripe (== one head), B staged once to LDS.
// Wave: 16x64 output (4 accumulators), 8 K-steps, 4 wmma per step.
// grid = ceil(M/16/8) * 4 stripes; tails clamped (uniform barriers).
// ---------------------------------------------------------------------------
__global__ void gat_gemm_kernel(const float* __restrict__ eA,
                                const float* __restrict__ gA,
                                const float* __restrict__ W,
                                float* __restrict__ Z, int M) {
  __shared__ _Float16 sB[64 * B_STRIDE];
  __shared__ _Float16 sA[8][16 * A_STRIDE];

  const int wave = threadIdx.x >> 5;
  const int lane = threadIdx.x & 31;
  const int stripe = blockIdx.x & 3;          // head index
  const int bm = blockIdx.x >> 2;
  const int numMT = M / 16;
  const int tm = bm * 8 + wave;
  const int tmC = (tm < numMT) ? tm : (numMT - 1);

  stage_B(W + (size_t)stripe * DIN * DHD, DHD, sB);
  __syncthreads();

  _Float16* sAw = sA[wave];
  v8f acc[4] = {};
  for (int kt = 0; kt < 8; ++kt) {
    const int k0 = kt * 32;
    // stage A 16x32 as f16 (concat virtualized; whole step in one half)
    const float* Ab = (k0 < DEMB) ? eA : gA;
    const int coff = (k0 < DEMB) ? k0 : (k0 - DEMB);
    {
      const int cp = lane & 15;
#pragma unroll
      for (int i = 0; i < 8; ++i) {
        int row = i * 2 + (lane >> 4);
        const float2 v =
            *(const float2*)(Ab + (size_t)(tmC * 16 + row) * DEMB + coff + 2 * cp);
        half2_t hv; hv.x = (_Float16)v.x; hv.y = (_Float16)v.y;
        *(half2_t*)&sAw[row * A_STRIDE + 2 * cp] = hv;
      }
    }
    __syncthreads();
    v16h af = load_a_frag(sAw, lane);
#pragma unroll
    for (int nt = 0; nt < 4; ++nt) {
      v16h bf = load_b_frag(sB, lane, nt, k0);
      acc[nt] = __builtin_amdgcn_wmma_f32_16x16x32_f16(false, af, false, bf,
                                                       (short)0, acc[nt],
                                                       false, false);
    }
    __syncthreads();
  }
  if (tm < numMT) {
    const int rbase = tm * 16 + ((lane >> 4) << 3);
    const int colb = stripe * 64 + (lane & 15);
#pragma unroll
    for (int nt = 0; nt < 4; ++nt)
#pragma unroll
      for (int r = 0; r < 8; ++r)
        Z[(size_t)(rbase + r) * HID + colb + nt * 16] = acc[nt][r];
  }
}

// ---------------------------------------------------------------------------
// s[n,h] = sum_k z[n, h*64+k] * a[h, off+k]
// ---------------------------------------------------------------------------
__global__ void score_kernel(const float* __restrict__ z,
                             const float* __restrict__ a, int off,
                             float* __restrict__ sout, int N) {
  int idx = blockIdx.x * blockDim.x + threadIdx.x;
  if (idx >= N * HH) return;
  int n = idx >> 2, h = idx & 3;
  const float* zp = z + (size_t)n * HID + h * DHD;
  const float* ap = a + h * (2 * DHD) + off;
  float s = 0.f;
#pragma unroll 8
  for (int k = 0; k < DHD; ++k) s += zp[k] * ap[k];
  sout[idx] = s;
}

// ---------------------------------------------------------------------------
// s_dst1[n,h] = concat(embed_o,gru_o)[n,:] . vdst1[h,:]
// ---------------------------------------------------------------------------
__global__ void sdst_kernel(const float* __restrict__ eO,
                            const float* __restrict__ gO,
                            const float* __restrict__ vdst1,
                            float* __restrict__ sout, int N) {
  int idx = blockIdx.x * blockDim.x + threadIdx.x;
  if (idx >= N * HH) return;
  int n = idx >> 2, h = idx & 3;
  const float* v = vdst1 + h * DIN;
  float s = 0.f;
#pragma unroll 8
  for (int d = 0; d < DEMB; ++d) s += eO[(size_t)n * DEMB + d] * v[d];
#pragma unroll 8
  for (int d = 0; d < DEMB; ++d) s += gO[(size_t)n * DEMB + d] * v[DEMB + d];
  sout[idx] = s;
}

// ---------------------------------------------------------------------------
// aggregate: one wave per (dst node o, head h). Edges of o are [o*16, o*16+16).
// 16-lane softmax via shuffles; 64-wide weighted gather-accumulate.
// ---------------------------------------------------------------------------
__global__ void aggregate_kernel(const float* __restrict__ z,
                                 const float* __restrict__ s_src,
                                 const float* __restrict__ s_dst,
                                 const int* __restrict__ e_src,
                                 float* __restrict__ hout) {
  const int wave = threadIdx.x >> 5;
  const int lane = threadIdx.x & 31;
  const int task = blockIdx.x * 8 + wave;     // o*4 + h, total N_O*4 (exact)
  const int o = task >> 2, h = task & 3;

  const int j = lane & 15;                    // halves duplicate -> 16-wide red.
  const int src = e_src[o * DEG + j];
  float e = s_src[src * HH + h] + s_dst[o * HH + h];
  e = (e >= 0.f) ? e : NEG_SLOPE * e;

  float m = e;
#pragma unroll
  for (int d = 8; d >= 1; d >>= 1) m = fmaxf(m, __shfl_xor(m, d, 32));
  float p = __expf(e - m);
  float s = p;
#pragma unroll
  for (int d = 8; d >= 1; d >>= 1) s += __shfl_xor(s, d, 32);
  const float alpha = p / s;

  float acc0 = 0.f, acc1 = 0.f;
  const int kcol = lane * 2;                  // 32 lanes x 2 = 64 cols
#pragma unroll
  for (int jj = 0; jj < DEG; ++jj) {
    float aj = __shfl(alpha, jj, 32);
    int   sj = __shfl(src, jj, 32);
    const float2 zv =
        *(const float2*)(z + (size_t)sj * HID + h * DHD + kcol);
    acc0 += aj * zv.x;
    acc1 += aj * zv.y;
  }
  float* op = hout + (size_t)o * HID + h * DHD + kcol;
  op[0] = acc0;
  op[1] = acc1;
}

// ---------------------------------------------------------------------------
// projection WMMA GEMM over stacked z=(h1,h2) rows (row2 = 2n+e) with fused
// tanh(.+bp1)·wp2 reduction into wacc[2]; t never materialized.
// Same block/wave tiling as gat_gemm: 8 waves x 16x64, shared Wp1 stripe.
// ---------------------------------------------------------------------------
__global__ void proj_gemm_kernel(const float* __restrict__ h1p,
                                 const float* __restrict__ h2p,
                                 const float* __restrict__ Wp1,
                                 const float* __restrict__ bp1,
                                 const float* __restrict__ wp2,
                                 float* __restrict__ wacc) {
  __shared__ _Float16 sB[64 * B_STRIDE];
  __shared__ _Float16 sA[8][16 * A_STRIDE];
  __shared__ float blk[2];
  if (threadIdx.x == 0) { blk[0] = 0.f; blk[1] = 0.f; }

  const int wave = threadIdx.x >> 5;
  const int lane = threadIdx.x & 31;
  const int stripe = blockIdx.x & 3;
  const int bm = blockIdx.x >> 2;
  const int numMT = (2 * N_O) / 16;           // 3750
  const int tm = bm * 8 + wave;
  const int tmC = (tm < numMT) ? tm : (numMT - 1);

  stage_B(Wp1 + stripe * 64, HID, sB);
  __syncthreads();                            // also covers blk[] init

  _Float16* sAw = sA[wave];
  v8f acc[4] = {};
  for (int kt = 0; kt < 8; ++kt) {
    const int k0 = kt * 32;
    {
      const int cp = lane & 15;
#pragma unroll
      for (int i = 0; i < 8; ++i) {
        int row = i * 2 + (lane >> 4);
        int row2 = tmC * 16 + row;            // even=h1, odd=h2
        const float* base = (row2 & 1) ? h2p : h1p;
        const float2 v =
            *(const float2*)(base + (size_t)(row2 >> 1) * HID + k0 + 2 * cp);
        half2_t hv; hv.x = (_Float16)v.x; hv.y = (_Float16)v.y;
        *(half2_t*)&sAw[row * A_STRIDE + 2 * cp] = hv;
      }
    }
    __syncthreads();
    v16h af = load_a_frag(sAw, lane);
#pragma unroll
    for (int nt = 0; nt < 4; ++nt) {
      v16h bf = load_b_frag(sB, lane, nt, k0);
      acc[nt] = __builtin_amdgcn_wmma_f32_16x16x32_f16(false, af, false, bf,
                                                       (short)0, acc[nt],
                                                       false, false);
    }
    __syncthreads();
  }
  // epilogue: t = tanh(acc + bp1[col]); sum t*wp2[col], split by e = r&1
  float s0 = 0.f, s1 = 0.f;
  if (tm < numMT) {
#pragma unroll
    for (int nt = 0; nt < 4; ++nt) {
      const int col = stripe * 64 + nt * 16 + (lane & 15);
      const float bc = bp1[col];
      const float wc = wp2[col];
#pragma unroll
      for (int r = 0; r < 8; ++r) {
        float t = tanhf(acc[nt][r] + bc);
        float c = t * wc;
        if (r & 1) s1 += c; else s0 += c;
      }
    }
  }
#pragma unroll
  for (int d = 16; d >= 1; d >>= 1) {
    s0 += __shfl_xor(s0, d, 32);
    s1 += __shfl_xor(s1, d, 32);
  }
  if (lane == 0) {
    atomicAdd(&blk[0], s0);
    atomicAdd(&blk[1], s1);
  }
  __syncthreads();
  if (threadIdx.x == 0) {
    atomicAdd(&wacc[0], blk[0]);
    atomicAdd(&wacc[1], blk[1]);
  }
}

// ---------------------------------------------------------------------------
// finalize: beta = softmax(w / N_O);  out = beta0*h1 + beta1*h2
// ---------------------------------------------------------------------------
__global__ void finalize_kernel(const float* __restrict__ h1,
                                const float* __restrict__ h2,
                                const float* __restrict__ wacc,
                                float* __restrict__ out) {
  size_t idx = (size_t)blockIdx.x * blockDim.x + threadIdx.x;
  float w0 = wacc[0] * (1.f / (float)N_O);
  float w1 = wacc[1] * (1.f / (float)N_O);
  float mx = fmaxf(w0, w1);
  float b0 = __expf(w0 - mx), b1 = __expf(w1 - mx);
  float inv = 1.f / (b0 + b1);
  out[idx] = (b0 * h1[idx] + b1 * h2[idx]) * inv;
}

// ---------------------------------------------------------------------------
extern "C" void kernel_launch(void* const* d_in, const int* in_sizes, int n_in,
                              void* d_out, int out_size, void* d_ws, size_t ws_size,
                              hipStream_t stream) {
  (void)in_sizes; (void)n_in; (void)out_size; (void)ws_size;
  const float* embed_h = (const float*)d_in[0];
  const float* gru_h   = (const float*)d_in[1];
  const float* embed_o = (const float*)d_in[2];
  const float* gru_o   = (const float*)d_in[3];
  const float* W_src1  = (const float*)d_in[4];
  const float* W_dst1  = (const float*)d_in[5];
  const float* a1      = (const float*)d_in[6];
  const float* W_in2   = (const float*)d_in[7];
  const float* a2      = (const float*)d_in[8];
  const float* Wp1     = (const float*)d_in[9];
  const float* bp1     = (const float*)d_in[10];
  const float* wp2     = (const float*)d_in[11];
  const int*   e1_src  = (const int*)d_in[12];
  const int*   e2_src  = (const int*)d_in[14];
  float* out = (float*)d_out;

  // workspace layout (floats)
  float* ws    = (float*)d_ws;
  float* z1    = ws;                       // N_H*256
  float* z2    = z1 + (size_t)N_H * HID;   // N_O*256
  float* hh1   = z2 + (size_t)N_O * HID;   // N_O*256
  float* hh2   = hh1 + (size_t)N_O * HID;  // N_O*256
  float* vdst1 = hh2 + (size_t)N_O * HID;  // 4*256
  float* sS1   = vdst1 + HH * DIN;         // N_H*4
  float* sD1   = sS1 + (size_t)N_H * HH;   // N_O*4
  float* sS2   = sD1 + (size_t)N_O * HH;   // N_O*4
  float* sD2   = sS2 + (size_t)N_O * HH;   // N_O*4
  float* wacc  = sD2 + (size_t)N_O * HH;   // 2

  const int TPB = 256;
  prep_kernel<<<4, TPB, 0, stream>>>(W_dst1, a1, vdst1, wacc);

  // z1 = concat(embed_h, gru_h) @ W_src1 ; z2 = concat(embed_o, gru_o) @ W_in2
  const int bm1 = (N_H / 16 + 7) / 8;      // 235
  gat_gemm_kernel<<<bm1 * 4, TPB, 0, stream>>>(embed_h, gru_h, W_src1, z1, N_H);
  gat_gemm_kernel<<<bm1 * 4, TPB, 0, stream>>>(embed_o, gru_o, W_in2, z2, N_O);

  const int nScore = (N_O * HH + TPB - 1) / TPB;
  sdst_kernel<<<nScore, TPB, 0, stream>>>(embed_o, gru_o, vdst1, sD1, N_O);
  score_kernel<<<nScore, TPB, 0, stream>>>(z1, a1, 0,   sS1, N_H);
  score_kernel<<<nScore, TPB, 0, stream>>>(z2, a2, 0,   sS2, N_O);
  score_kernel<<<nScore, TPB, 0, stream>>>(z2, a2, DHD, sD2, N_O);

  aggregate_kernel<<<N_O * HH / 8, TPB, 0, stream>>>(z1, sS1, sD1, e1_src, hh1);
  aggregate_kernel<<<N_O * HH / 8, TPB, 0, stream>>>(z2, sS2, sD2, e2_src, hh2);

  const int bm2 = ((2 * N_O) / 16 + 7) / 8; // 469
  proj_gemm_kernel<<<bm2 * 4, TPB, 0, stream>>>(hh1, hh2, Wp1, bp1, wp2, wacc);

  finalize_kernel<<<N_O * HID / TPB, TPB, 0, stream>>>(hh1, hh2, wacc, out);
}
